// GeoPoseNet_6700148982571
// MI455X (gfx1250) — matrix operand order repaired and checked by reading, not compiled, not run on hardware
//
#include <hip/hip_runtime.h>
#include <hip/hip_bf16.h>
#include <math.h>

#define N_NODES   40000
#define N_EDGES   640000
#define N_GRAPHS  400
#define D1        256     // conv1 out (4 heads * 64), also conv2 in
#define C2        128     // conv2 out channels (post head-mean)
#define NCLS      12

typedef __attribute__((ext_vector_type(16))) _Float16 v16h;
typedef __attribute__((ext_vector_type(8)))  float    v8f;

union Frag16 { v16h h; unsigned int u[8]; };

__device__ inline float waveSum(float v) {
#pragma unroll
  for (int off = 16; off > 0; off >>= 1) v += __shfl_xor(v, off, 32);
  return v;
}

// float atomic max via int(+)/uint(-) ordering trick; init value must be -inf
__device__ inline void atomicMaxF(float* addr, float v) {
  if (v >= 0.0f) atomicMax((int*)addr, __float_as_int(v));
  else           atomicMin((unsigned int*)addr, (unsigned int)__float_as_int(v));
}

// ---------------- conv1: node transforms (K=2, pure streaming) ----------------
__global__ void k_node1(const float* __restrict__ x,
                        const float* __restrict__ Wl, const float* __restrict__ bl,
                        const float* __restrict__ Wr, const float* __restrict__ br,
                        float* __restrict__ xl1, float* __restrict__ xr1) {
  int id = blockIdx.x * blockDim.x + threadIdx.x;
  if (id >= N_NODES * D1) return;
  int n = id >> 8, c = id & 255;
  float x0 = x[2 * n], x1 = x[2 * n + 1];
  xl1[id] = fmaf(x0, Wl[c], fmaf(x1, Wl[D1 + c], bl[c]));
  xr1[id] = fmaf(x0, Wr[c], fmaf(x1, Wr[D1 + c], br[c]));
}

__global__ void k_init1(const float* __restrict__ bias1,
                        float* __restrict__ agg, float* __restrict__ nmax,
                        float* __restrict__ nden) {
  int id = blockIdx.x * blockDim.x + threadIdx.x;
  if (id >= N_NODES * D1) return;
  int n = id >> 8, c = id & 255;
  agg[id] = bias1[c];
  if (c < 4) { nmax[n * 4 + c] = -__builtin_inff(); nden[n * 4 + c] = 0.0f; }
}

// wave per (edge, head); 32 lanes cover 64 channels (2 each)
__global__ void k_elog1(const int* __restrict__ ei, const float* __restrict__ ea,
                        const float* __restrict__ We, const float* __restrict__ att,
                        const float* __restrict__ xl1, const float* __restrict__ xr1,
                        float* __restrict__ eatt, float* __restrict__ nmax) {
  int wid  = (blockIdx.x * blockDim.x + threadIdx.x) >> 5;
  int lane = threadIdx.x & 31;
  if (wid >= N_EDGES * 4) return;
  int e = wid >> 2, h = wid & 3;
  int src = ei[e], dst = ei[N_EDGES + e];
  float a0 = ea[4 * e], a1 = ea[4 * e + 1], a2 = ea[4 * e + 2], a3 = ea[4 * e + 3];
  const float* xls = xl1 + (size_t)src * D1 + h * 64;
  const float* xrd = xr1 + (size_t)dst * D1 + h * 64;
  const float* Wc  = We + h * 64;
  float s = 0.0f;
#pragma unroll
  for (int i = 0; i < 2; ++i) {
    int c = lane + 32 * i;
    float ee = fmaf(a0, Wc[c], fmaf(a1, Wc[D1 + c], fmaf(a2, Wc[2 * D1 + c], a3 * Wc[3 * D1 + c])));
    float z = xls[c] + xrd[c] + ee;
    z = (z > 0.0f) ? z : 0.2f * z;              // leaky_relu 0.2
    s = fmaf(z, att[h * 64 + c], s);
  }
  s = waveSum(s);
  if (lane == 0) { eatt[wid] = s; atomicMaxF(&nmax[dst * 4 + h], s); }
}

__global__ void k_eexp1(const int* __restrict__ ei, float* __restrict__ eatt,
                        const float* __restrict__ nmax, float* __restrict__ nden) {
  int id = blockIdx.x * blockDim.x + threadIdx.x;
  if (id >= N_EDGES * 4) return;
  int e = id >> 2, h = id & 3;
  int dst = ei[N_EDGES + e];
  float ex = expf(eatt[id] - nmax[dst * 4 + h]);
  eatt[id] = ex;
  atomicAdd(&nden[dst * 4 + h], ex);
}

__global__ void k_eagg1(const int* __restrict__ ei, const float* __restrict__ eatt,
                        const float* __restrict__ nden, const float* __restrict__ xl1,
                        float* __restrict__ agg) {
  int wid  = (blockIdx.x * blockDim.x + threadIdx.x) >> 5;
  int lane = threadIdx.x & 31;
  if (wid >= N_EDGES * 4) return;
  int e = wid >> 2, h = wid & 3;
  int src = ei[e], dst = ei[N_EDGES + e];
  float alpha = eatt[wid] / (nden[dst * 4 + h] + 1e-16f);
  const float* xls = xl1 + (size_t)src * D1 + h * 64;
  float* out = agg + (size_t)dst * D1 + h * 64;
#pragma unroll
  for (int i = 0; i < 2; ++i) {
    int c = lane + 32 * i;
    atomicAdd(&out[c], xls[c] * alpha);
  }
}

// ---------------- batchnorm stats (one block per channel) ----------------
__global__ void k_bnstats(const float* __restrict__ X, int nch,
                          float* __restrict__ mean, float* __restrict__ rstd) {
  int c = blockIdx.x;
  float s = 0.0f, s2 = 0.0f;
  for (int n = threadIdx.x; n < N_NODES; n += blockDim.x) {
    float v = X[(size_t)n * nch + c];
    s += v; s2 = fmaf(v, v, s2);
  }
  __shared__ float sh[256], sh2[256];
  sh[threadIdx.x] = s; sh2[threadIdx.x] = s2;
  __syncthreads();
  for (int off = blockDim.x >> 1; off > 0; off >>= 1) {
    if ((int)threadIdx.x < off) { sh[threadIdx.x] += sh[threadIdx.x + off]; sh2[threadIdx.x] += sh2[threadIdx.x + off]; }
    __syncthreads();
  }
  if (threadIdx.x == 0) {
    float m = sh[0] / (float)N_NODES;
    float var = sh2[0] / (float)N_NODES - m * m;
    mean[c] = m;
    rstd[c] = rsqrtf(var + 1e-5f);
  }
}

// BN + ELU, emit f16 for the WMMA GEMM
__global__ void k_bn_elu_f16(const float* __restrict__ X, const float* __restrict__ mean,
                             const float* __restrict__ rstd, const float* __restrict__ g,
                             const float* __restrict__ be, _Float16* __restrict__ H) {
  int id = blockIdx.x * blockDim.x + threadIdx.x;
  if (id >= N_NODES * D1) return;
  int c = id & 255;
  float v = fmaf((X[id] - mean[c]) * rstd[c], g[c], be[c]);
  v = (v > 0.0f) ? v : expm1f(v);
  H[id] = (_Float16)v;
}

// ---------------- pack [Wl2|Wr2] -> per-lane WMMA B fragments ----------------
// B 32x16 f16, this code's contiguous k order (not the ISA table order): lane gives N=lane&15; lanes 0-15 hold K=0..15, lanes 16-31 K=16..31
__global__ void k_packW(const float* __restrict__ Wl2, const float* __restrict__ Wr2,
                        _Float16* __restrict__ Bp) {
  int t = blockIdx.x * blockDim.x + threadIdx.x;  // (ntile,kk,lane)
  if (t >= 32 * 8 * 32) return;
  int lane = t & 31, kk = (t >> 5) & 7, ntile = t >> 8;
  const float* W = (ntile < 16) ? Wl2 : Wr2;
  int ncol = (ntile & 15) * 16 + (lane & 15);
  int kbase = kk * 32 + (lane >> 4) * 16;
  _Float16* dst = Bp + (size_t)t * 16;
#pragma unroll
  for (int j = 0; j < 16; ++j) dst[j] = (_Float16)W[(size_t)(kbase + j) * D1 + ncol];
}

// ---------------- conv2 fused dual GEMM: [xl2|xr2] = h1f16 @ [Wl2|Wr2] ----------------
// one wave -> one 16x16 tile; 8 x v_wmma_f32_16x16x32_f16 over K=256
__global__ void k_gemm2(const _Float16* __restrict__ Ah, const _Float16* __restrict__ Bp,
                        const float* __restrict__ bl2, const float* __restrict__ br2,
                        float* __restrict__ xl2, float* __restrict__ xr2) {
  int lane  = threadIdx.x;                         // 0..31
  int ntile = blockIdx.y * blockDim.y + threadIdx.y; // 0..31 (512 cols)
  int mtile = blockIdx.x;                          // 0..2499
  int half  = lane >> 4;
  // A 16x32 f16, this code's contiguous k order: M=lane&15; lanes 0-15 hold K=0..15, lanes 16-31 K=16..31 (matches packed B)
  int arow = mtile * 16 + (lane & 15);
  const unsigned int* A32 = (const unsigned int*)(Ah + (size_t)arow * D1);
  const unsigned int* B32 = (const unsigned int*)Bp;
  v8f acc = {};
#pragma unroll
  for (int kk = 0; kk < 8; ++kk) {
    Frag16 a, b;
    int abase = kk * 16 + half * 8;                // dword index in row
#pragma unroll
    for (int p = 0; p < 4; ++p) { a.u[p] = A32[abase + p]; a.u[4 + p] = A32[abase + 4 + p]; }
    const unsigned int* bp = B32 + ((size_t)(ntile * 8 + kk) * 32 + lane) * 8;
#pragma unroll
    for (int p = 0; p < 8; ++p) b.u[p] = bp[p];
    acc = __builtin_amdgcn_wmma_f32_16x16x32_f16(false, a.h, false, b.h,
                                                 (short)0, acc, false, false);
  }
  // D layout: N=lane&15, M = 8*(lane>>4) + r
  int nloc = (ntile & 15) * 16 + (lane & 15);
  float* C = (ntile < 16) ? xl2 : xr2;
  float bias = ((ntile < 16) ? bl2 : br2)[nloc];
  int mbase = mtile * 16 + 8 * half;
#pragma unroll
  for (int r = 0; r < 8; ++r)
    C[(size_t)(mbase + r) * D1 + nloc] = acc[r] + bias;
}

// ---------------- conv2 edge passes (H=2, C=128; 4 channels per lane) ----------------
__global__ void k_init2(float* __restrict__ agg, float* __restrict__ nmax,
                        float* __restrict__ nden) {
  int id = blockIdx.x * blockDim.x + threadIdx.x;
  if (id >= N_NODES * D1) return;
  int n = id >> 8, c = id & 255;
  agg[id] = 0.0f;
  if (c < 2) { nmax[n * 2 + c] = -__builtin_inff(); nden[n * 2 + c] = 0.0f; }
}

__global__ void k_elog2(const int* __restrict__ ei, const float* __restrict__ ea,
                        const float* __restrict__ We, const float* __restrict__ att,
                        const float* __restrict__ xl2, const float* __restrict__ xr2,
                        float* __restrict__ eatt, float* __restrict__ nmax) {
  int wid  = (blockIdx.x * blockDim.x + threadIdx.x) >> 5;
  int lane = threadIdx.x & 31;
  if (wid >= N_EDGES * 2) return;
  int e = wid >> 1, h = wid & 1;
  int src = ei[e], dst = ei[N_EDGES + e];
  float a0 = ea[4 * e], a1 = ea[4 * e + 1], a2 = ea[4 * e + 2], a3 = ea[4 * e + 3];
  const float* xls = xl2 + (size_t)src * D1 + h * C2;
  const float* xrd = xr2 + (size_t)dst * D1 + h * C2;
  const float* Wc  = We + h * C2;
  float s = 0.0f;
#pragma unroll
  for (int i = 0; i < 4; ++i) {
    int c = lane + 32 * i;
    float ee = fmaf(a0, Wc[c], fmaf(a1, Wc[D1 + c], fmaf(a2, Wc[2 * D1 + c], a3 * Wc[3 * D1 + c])));
    float z = xls[c] + xrd[c] + ee;
    z = (z > 0.0f) ? z : 0.2f * z;
    s = fmaf(z, att[h * C2 + c], s);
  }
  s = waveSum(s);
  if (lane == 0) { eatt[wid] = s; atomicMaxF(&nmax[dst * 2 + h], s); }
}

__global__ void k_eexp2(const int* __restrict__ ei, float* __restrict__ eatt,
                        const float* __restrict__ nmax, float* __restrict__ nden) {
  int id = blockIdx.x * blockDim.x + threadIdx.x;
  if (id >= N_EDGES * 2) return;
  int e = id >> 1, h = id & 1;
  int dst = ei[N_EDGES + e];
  float ex = expf(eatt[id] - nmax[dst * 2 + h]);
  eatt[id] = ex;
  atomicAdd(&nden[dst * 2 + h], ex);
}

__global__ void k_eagg2(const int* __restrict__ ei, const float* __restrict__ eatt,
                        const float* __restrict__ nden, const float* __restrict__ xl2,
                        float* __restrict__ agg) {
  int wid  = (blockIdx.x * blockDim.x + threadIdx.x) >> 5;
  int lane = threadIdx.x & 31;
  if (wid >= N_EDGES * 2) return;
  int e = wid >> 1, h = wid & 1;
  int src = ei[e], dst = ei[N_EDGES + e];
  float alpha = eatt[wid] / (nden[dst * 2 + h] + 1e-16f);
  const float* xls = xl2 + (size_t)src * D1 + h * C2;
  float* out = agg + (size_t)dst * D1 + h * C2;
#pragma unroll
  for (int i = 0; i < 4; ++i) {
    int c = lane + 32 * i;
    atomicAdd(&out[c], xls[c] * alpha);
  }
}

// mean over 2 heads + bias2
__global__ void k_hmean(const float* __restrict__ agg, const float* __restrict__ bias2,
                        float* __restrict__ h2) {
  int id = blockIdx.x * blockDim.x + threadIdx.x;
  if (id >= N_NODES * C2) return;
  int n = id >> 7, c = id & 127;
  h2[id] = 0.5f * (agg[(size_t)n * D1 + c] + agg[(size_t)n * D1 + C2 + c]) + bias2[c];
}

__global__ void k_bn_elu_f32(float* __restrict__ X, const float* __restrict__ mean,
                             const float* __restrict__ rstd, const float* __restrict__ g,
                             const float* __restrict__ be) {
  int id = blockIdx.x * blockDim.x + threadIdx.x;
  if (id >= N_NODES * C2) return;
  int c = id & 127;
  float v = fmaf((X[id] - mean[c]) * rstd[c], g[c], be[c]);
  X[id] = (v > 0.0f) ? v : expm1f(v);
}

// ---------------- global mean pool + MLP head ----------------
__global__ void k_poolinit(float* __restrict__ pool, float* __restrict__ cnt) {
  int id = blockIdx.x * blockDim.x + threadIdx.x;
  if (id < N_GRAPHS * C2) pool[id] = 0.0f;
  if (id < N_GRAPHS) cnt[id] = 0.0f;
}

__global__ void k_pool(const int* __restrict__ batch, const float* __restrict__ h2,
                       float* __restrict__ pool, float* __restrict__ cnt) {
  int id = blockIdx.x * blockDim.x + threadIdx.x;
  if (id >= N_NODES * C2) return;
  int n = id >> 7, c = id & 127;
  int g = batch[n];
  atomicAdd(&pool[(size_t)g * C2 + c], h2[id]);
  if (c == 0) atomicAdd(&cnt[g], 1.0f);
}

__global__ void k_head(const float* __restrict__ pool, const float* __restrict__ cnt,
                       const float* __restrict__ W1, const float* __restrict__ b1,
                       const float* __restrict__ W2, const float* __restrict__ b2,
                       float* __restrict__ out) {
  int g = blockIdx.x;
  __shared__ float hin[C2];
  __shared__ float hm[64];
  float inv = 1.0f / fmaxf(cnt[g], 1.0f);
  for (int c = threadIdx.x; c < C2; c += 64) hin[c] = pool[(size_t)g * C2 + c] * inv;
  __syncthreads();
  int t = threadIdx.x;  // 0..63
  float s = b1[t];
  for (int i = 0; i < C2; ++i) s = fmaf(hin[i], W1[i * 64 + t], s);
  hm[t] = (s > 0.0f) ? s : expm1f(s);
  __syncthreads();
  if (t < NCLS) {
    float o = b2[t];
    for (int i = 0; i < 64; ++i) o = fmaf(hm[i], W2[i * NCLS + t], o);
    out[g * NCLS + t] = o;
  }
}

extern "C" void kernel_launch(void* const* d_in, const int* in_sizes, int n_in,
                              void* d_out, int out_size, void* d_ws, size_t ws_size,
                              hipStream_t stream) {
  const float* x     = (const float*)d_in[0];
  const int*   ei    = (const int*)d_in[1];
  const float* ea    = (const float*)d_in[2];
  const int*   batch = (const int*)d_in[3];
  const float* Wl1 = (const float*)d_in[4],  *bl1 = (const float*)d_in[5];
  const float* Wr1 = (const float*)d_in[6],  *br1 = (const float*)d_in[7];
  const float* We1 = (const float*)d_in[8],  *att1 = (const float*)d_in[9];
  const float* bias1 = (const float*)d_in[10], *g1 = (const float*)d_in[11], *be1 = (const float*)d_in[12];
  const float* Wl2 = (const float*)d_in[13], *bl2 = (const float*)d_in[14];
  const float* Wr2 = (const float*)d_in[15], *br2 = (const float*)d_in[16];
  const float* We2 = (const float*)d_in[17], *att2 = (const float*)d_in[18];
  const float* bias2 = (const float*)d_in[19], *g2 = (const float*)d_in[20], *be2 = (const float*)d_in[21];
  const float* W_lin1 = (const float*)d_in[22], *b_lin1 = (const float*)d_in[23];
  const float* W_lin2 = (const float*)d_in[24], *b_lin2 = (const float*)d_in[25];
  float* out = (float*)d_out;

  char* ws = (char*)d_ws;
  size_t off = 0;
  auto carve = [&](size_t bytes) -> char* {
    char* p = ws + off;
    off += (bytes + 255) & ~(size_t)255;
    return p;
  };
  float*    xl1  = (float*)carve((size_t)N_NODES * D1 * 4);   // reused as xl2
  float*    xr1  = (float*)carve((size_t)N_NODES * D1 * 4);   // reused as xr2
  float*    agg  = (float*)carve((size_t)N_NODES * D1 * 4);   // conv1 then conv2 aggregation
  float*    eatt = (float*)carve((size_t)N_EDGES * 4 * 4);    // conv2 reuses first half
  float*    nmax = (float*)carve((size_t)N_NODES * 4 * 4);
  float*    nden = (float*)carve((size_t)N_NODES * 4 * 4);
  float*    stats = (float*)carve(1024 * 4);                  // mean1|rstd1|mean2|rstd2
  char*     regH = carve((size_t)N_NODES * D1 * 2);           // h1 f16, then h2 f32 (same bytes)
  _Float16* h1h  = (_Float16*)regH;
  float*    h2   = (float*)regH;
  _Float16* wpack = (_Float16*)carve((size_t)32 * 8 * 32 * 16 * 2);
  float*    pool = (float*)carve((size_t)N_GRAPHS * C2 * 4);
  float*    cnt  = (float*)carve((size_t)N_GRAPHS * 4);
  float* mean1 = stats, *rstd1 = stats + 256, *mean2 = stats + 512, *rstd2 = stats + 640;

  const int TPB = 256;
  int gNC1 = (N_NODES * D1 + TPB - 1) / TPB;       // 40000*256 threads
  int gNC2 = (N_NODES * C2 + TPB - 1) / TPB;
  int gE4w = N_EDGES * 4 / 8;                      // wave-per-(e,h), 8 waves/block
  int gE2w = N_EDGES * 2 / 8;
  int gE4t = (N_EDGES * 4 + TPB - 1) / TPB;
  int gE2t = (N_EDGES * 2 + TPB - 1) / TPB;

  // ---- conv1 ----
  k_node1<<<gNC1, TPB, 0, stream>>>(x, Wl1, bl1, Wr1, br1, xl1, xr1);
  k_init1<<<gNC1, TPB, 0, stream>>>(bias1, agg, nmax, nden);
  k_elog1<<<gE4w, TPB, 0, stream>>>(ei, ea, We1, att1, xl1, xr1, eatt, nmax);
  k_eexp1<<<gE4t, TPB, 0, stream>>>(ei, eatt, nmax, nden);
  k_eagg1<<<gE4w, TPB, 0, stream>>>(ei, eatt, nden, xl1, agg);
  k_bnstats<<<D1, TPB, 0, stream>>>(agg, D1, mean1, rstd1);
  k_bn_elu_f16<<<gNC1, TPB, 0, stream>>>(agg, mean1, rstd1, g1, be1, h1h);

  // ---- conv2 transforms via WMMA ----
  k_packW<<<(32 * 8 * 32 + TPB - 1) / TPB, TPB, 0, stream>>>(Wl2, Wr2, wpack);
  {
    dim3 grid(N_NODES / 16, 8);   // 2500 x 8, 4 N-tiles per block
    dim3 blk(32, 4);
    k_gemm2<<<grid, blk, 0, stream>>>(h1h, wpack, bl2, br2, xl1, xr1);  // xl1/xr1 now hold xl2/xr2
  }

  // ---- conv2 attention + aggregation ----
  k_init2<<<gNC1, TPB, 0, stream>>>(agg, nmax, nden);
  k_elog2<<<gE2w, TPB, 0, stream>>>(ei, ea, We2, att2, xl1, xr1, eatt, nmax);
  k_eexp2<<<gE2t, TPB, 0, stream>>>(ei, eatt, nmax, nden);
  k_eagg2<<<gE2w, TPB, 0, stream>>>(ei, eatt, nden, xl1, agg);
  k_hmean<<<gNC2, TPB, 0, stream>>>(agg, bias2, h2);
  k_bnstats<<<C2, TPB, 0, stream>>>(h2, C2, mean2, rstd2);
  k_bn_elu_f32<<<gNC2, TPB, 0, stream>>>(h2, mean2, rstd2, g2, be2);

  // ---- pool + MLP head ----
  k_poolinit<<<(N_GRAPHS * C2 + TPB - 1) / TPB, TPB, 0, stream>>>(pool, cnt);
  k_pool<<<gNC2, TPB, 0, stream>>>(batch, h2, pool, cnt);
  k_head<<<N_GRAPHS, 64, 0, stream>>>(pool, cnt, W_lin1, b_lin1, W_lin2, b_lin2, out);
}
